// GConv_v0_27736898798375
// MI455X (gfx1250) — matrix operand order repaired and checked by reading, not compiled, not run on hardware
//
#include <hip/hip_runtime.h>
#include <hip/hip_bf16.h>
#include <stdint.h>

#define FEAT 256
#define MAXDEG 16

typedef __attribute__((ext_vector_type(16))) __bf16 v16bf;
typedef __attribute__((ext_vector_type(8)))  float  v8f;

union Frag { uint4 q[2]; v16bf v; };   // 16 bf16 elements = 32 bytes

// split f32 into truncated-bf16 hi and bf16(residual) lo; hi+lo ~ f (2^-16 rel)
__device__ __forceinline__ void split_bf16(float f, unsigned short& hi,
                                           unsigned short& lo) {
    unsigned u = __float_as_uint(f);
    hi = (unsigned short)(u >> 16);
    float r = f - __uint_as_float(u & 0xffff0000u);
    lo = (unsigned short)(__float_as_uint(r) >> 16);
}

// ---------------------------------------------------------------------------
// Stage 0: mean aggregation over neighbors. one block per node, one thread per
// feature column; neighbor rows are read coalesced per wave.
// ---------------------------------------------------------------------------
__global__ void k_aggregate(const float* __restrict__ x,
                            const int* __restrict__ nbr,
                            float* __restrict__ h0) {
    const int i = blockIdx.x;
    const int c = threadIdx.x;
    float s = 0.f;
    int cnt = 0;
#pragma unroll
    for (int j = 0; j < MAXDEG; ++j) {
        int id = nbr[i * MAXDEG + j];
        if (id >= 0) { s += x[(size_t)id * FEAT + c]; ++cnt; }
    }
    float inv = (cnt > 0) ? (1.0f / (float)cnt) : 0.0f;
    h0[(size_t)i * FEAT + c] = s * inv;
}

// ---------------------------------------------------------------------------
__global__ void k_init(unsigned int* __restrict__ cmin,
                       float* __restrict__ colsum,
                       float* __restrict__ colsq) {
    int t = threadIdx.x;
    cmin[t]   = 0xFFFFFFFFu;
    colsum[t] = 0.f;
    colsq[t]  = 0.f;
}

__device__ __forceinline__ unsigned f2key(float f) {
    unsigned u = __float_as_uint(f);
    return (u & 0x80000000u) ? ~u : (u | 0x80000000u);
}
__device__ __forceinline__ float key2f(unsigned k) {
    unsigned u = (k & 0x80000000u) ? (k & 0x7FFFFFFFu) : ~k;
    return __uint_as_float(u);
}

// ---------------------------------------------------------------------------
// Column-wise min of h0 (the "dummy row") via order-preserving uint keys.
// ---------------------------------------------------------------------------
__global__ void k_colmin(const float* __restrict__ h0,
                         unsigned int* __restrict__ cmin, int nrows) {
    const int c = threadIdx.x;
    unsigned best = 0xFFFFFFFFu;
    for (int r = blockIdx.x; r < nrows; r += gridDim.x) {
        unsigned k = f2key(h0[(size_t)r * FEAT + c]);
        best = best < k ? best : k;
    }
    atomicMin(&cmin[c], best);
}

// ---------------------------------------------------------------------------
// h1 = h0 + alpha * gather_max(h0, dummy) + (1+eps) * x,
// emitted as split-bf16 hi/lo row-major arrays (GEMM A operand format).
// ---------------------------------------------------------------------------
__global__ void k_combine(const float* __restrict__ x,
                          const int* __restrict__ nbr,
                          const float* __restrict__ h0,
                          const unsigned int* __restrict__ cmin,
                          const float* __restrict__ alpha_p,
                          const float* __restrict__ eps_p,
                          unsigned short* __restrict__ h1hi,
                          unsigned short* __restrict__ h1lo) {
    const int i = blockIdx.x;
    const int c = threadIdx.x;
    const float dummy = key2f(cmin[c]);
    float m = -3.402823466e+38f;
#pragma unroll
    for (int j = 0; j < MAXDEG; ++j) {
        int id = nbr[i * MAXDEG + j];
        float v = (id >= 0) ? h0[(size_t)id * FEAT + c] : dummy;
        m = fmaxf(m, v);
    }
    const float alpha = alpha_p[0];
    const float eps   = eps_p[0];
    float hv = h0[(size_t)i * FEAT + c];
    float v  = hv + alpha * m + (1.0f + eps) * x[(size_t)i * FEAT + c];
    unsigned short hi, lo;
    split_bf16(v, hi, lo);
    h1hi[(size_t)i * FEAT + c] = hi;
    h1lo[(size_t)i * FEAT + c] = lo;
}

// ---------------------------------------------------------------------------
// Split W (256x256 f32, row-major [k][n]) into bf16 hi/lo arrays stored in
// WMMA B-fragment order: element e of lane (h = lane>>4, n = ntile*16+lane&15)
// for (ntile,kstep) holds W[kstep*32 + 8h + e + (e>=8 ? 8 : 0)][n]   (ISA B layout).
// Storage index: ((ntile*8 + kstep)*32 + lane)*16 + e   (ushort).
// ---------------------------------------------------------------------------
__global__ void k_wsplit(const float* __restrict__ W,
                         unsigned short* __restrict__ hi,
                         unsigned short* __restrict__ lo) {
    int t = blockIdx.x * blockDim.x + threadIdx.x;   // 0..4095
    int lane  = t & 31;
    int frag  = t >> 5;                              // ntile*8 + kstep
    int kstep = frag & 7;
    int ntile = frag >> 3;
    int n     = ntile * 16 + (lane & 15);
    int kbase = kstep * 32 + ((lane >> 4) << 3);
    size_t ob = ((size_t)frag * 32 + lane) * 16;
#pragma unroll
    for (int e = 0; e < 16; ++e) {
        unsigned short h, l;
        split_bf16(W[(size_t)(kbase + e + (e >= 8 ? 8 : 0)) * FEAT + n], h, l);
        hi[ob + e] = h;
        lo[ob + e] = l;
    }
}

// ---------------------------------------------------------------------------
// GEMM: C = relu(A @ W + b).  A given pre-split as bf16 hi/lo row-major:
//   A@W ~= Ahi@Whi + Ahi@Wlo + Alo@Whi       (error ~2^-15 relative)
// Block = 8 waves; wave w owns rows [bx*128+16w, +16) x 32 cols.
// A fragment = two aligned b128 loads per operand (no VALU in the K loop).
// mode 0: store split-bf16 hi/lo (feeds next GEMM).
// mode 1: store f32 to outF + fused BatchNorm column sums.
// ---------------------------------------------------------------------------
__global__ void __launch_bounds__(256)
k_gemm(const unsigned short* __restrict__ Ahi,
       const unsigned short* __restrict__ Alo,
       const unsigned short* __restrict__ Whi,
       const unsigned short* __restrict__ Wlo,
       const float* __restrict__ bias,
       unsigned short* __restrict__ outHi,
       unsigned short* __restrict__ outLo,
       float* __restrict__ outF,
       int nrows, int mode,
       float* __restrict__ colsum, float* __restrict__ colsumsq) {
    __shared__ float ssum[32];
    __shared__ float ssq[32];

    const int tid   = threadIdx.x;
    const int wave  = tid >> 5;
    const int lane  = tid & 31;
    const int lhalf = lane >> 4;       // 0/1
    const int lmod  = lane & 15;

    const int rowBase = blockIdx.x * 128 + wave * 16;
    const int colBase = blockIdx.y * 32;
    const int nt0     = blockIdx.y * 2;       // ntile of first 16-col tile

    int arow = rowBase + lmod;                // A row held by this lane
    if (arow > nrows - 1) arow = nrows - 1;   // clamp (EXEC must stay full)
    const unsigned short* arh = Ahi + (size_t)arow * FEAT + 8 * lhalf;
    const unsigned short* arl = Alo + (size_t)arow * FEAT + 8 * lhalf;

    v8f acc0 = {};
    v8f acc1 = {};

#pragma unroll
    for (int ks = 0; ks < 8; ++ks) {
        // ---- A fragments (ISA 16-bit A 16x32 layout):
        //   e<8  -> K = ks*32 + 8*lhalf + e ;  e>=8 -> +8 more
        Frag ah, al;
        ah.q[0] = *(const uint4*)(arh + ks * 32);
        ah.q[1] = *(const uint4*)(arh + ks * 32 + 16);
        al.q[0] = *(const uint4*)(arl + ks * 32);
        al.q[1] = *(const uint4*)(arl + ks * 32 + 16);

        // ---- B fragments (pre-swizzled): two aligned 16B loads each.
        {
            size_t bidx = ((size_t)((nt0 + 0) * 8 + ks) * 32 + lane) * 2;
            Frag bh, bl;
            bh.q[0] = ((const uint4*)Whi)[bidx];
            bh.q[1] = ((const uint4*)Whi)[bidx + 1];
            bl.q[0] = ((const uint4*)Wlo)[bidx];
            bl.q[1] = ((const uint4*)Wlo)[bidx + 1];
            acc0 = __builtin_amdgcn_wmma_f32_16x16x32_bf16(false, ah.v, false, bh.v, (short)0, acc0, false, false);
            acc0 = __builtin_amdgcn_wmma_f32_16x16x32_bf16(false, ah.v, false, bl.v, (short)0, acc0, false, false);
            acc0 = __builtin_amdgcn_wmma_f32_16x16x32_bf16(false, al.v, false, bh.v, (short)0, acc0, false, false);
        }
        {
            size_t bidx = ((size_t)((nt0 + 1) * 8 + ks) * 32 + lane) * 2;
            Frag bh, bl;
            bh.q[0] = ((const uint4*)Whi)[bidx];
            bh.q[1] = ((const uint4*)Whi)[bidx + 1];
            bl.q[0] = ((const uint4*)Wlo)[bidx];
            bl.q[1] = ((const uint4*)Wlo)[bidx + 1];
            acc1 = __builtin_amdgcn_wmma_f32_16x16x32_bf16(false, ah.v, false, bh.v, (short)0, acc1, false, false);
            acc1 = __builtin_amdgcn_wmma_f32_16x16x32_bf16(false, ah.v, false, bl.v, (short)0, acc1, false, false);
            acc1 = __builtin_amdgcn_wmma_f32_16x16x32_bf16(false, al.v, false, bh.v, (short)0, acc1, false, false);
        }
    }

    // ---- epilogue: bias + ReLU (+ output in requested format).
    // C layout: VGPR r -> row = rowBase + r + 8*lhalf, col = tile + lmod.
    float s0 = 0.f, s20 = 0.f, s1 = 0.f, s21 = 0.f;
    const int col0 = colBase + lmod;
    const int col1 = colBase + 16 + lmod;
    const float b0  = bias[col0];
    const float b1v = bias[col1];
#pragma unroll
    for (int r = 0; r < 8; ++r) {
        int row = rowBase + r + lhalf * 8;
        if (row < nrows) {
            float v = fmaxf(acc0[r] + b0, 0.f);
            float w = fmaxf(acc1[r] + b1v, 0.f);
            if (mode == 0) {
                unsigned short h, l;
                split_bf16(v, h, l);
                outHi[(size_t)row * FEAT + col0] = h;
                outLo[(size_t)row * FEAT + col0] = l;
                split_bf16(w, h, l);
                outHi[(size_t)row * FEAT + col1] = h;
                outLo[(size_t)row * FEAT + col1] = l;
            } else {
                outF[(size_t)row * FEAT + col0] = v;
                outF[(size_t)row * FEAT + col1] = w;
                s0 += v; s20 += v * v;
                s1 += w; s21 += w * w;
            }
        }
    }

    if (mode == 1) {
        if (tid < 32) { ssum[tid] = 0.f; ssq[tid] = 0.f; }
        __syncthreads();
        atomicAdd(&ssum[lmod],      s0);
        atomicAdd(&ssq [lmod],      s20);
        atomicAdd(&ssum[16 + lmod], s1);
        atomicAdd(&ssq [16 + lmod], s21);
        __syncthreads();
        if (tid < 32) {
            atomicAdd(&colsum  [colBase + tid], ssum[tid]);
            atomicAdd(&colsumsq[colBase + tid], ssq [tid]);
        }
    }
}

// ---------------------------------------------------------------------------
// BatchNorm apply, in place over d_out.
// ---------------------------------------------------------------------------
__global__ void k_bn(float* __restrict__ out,
                     const float* __restrict__ colsum,
                     const float* __restrict__ colsumsq,
                     const float* __restrict__ gamma,
                     const float* __restrict__ beta, int nrows) {
    const size_t total = (size_t)nrows * FEAT;
    const float invN = 1.0f / (float)nrows;
    for (size_t idx = (size_t)blockIdx.x * blockDim.x + threadIdx.x;
         idx < total; idx += (size_t)gridDim.x * blockDim.x) {
        int c = (int)(idx & (FEAT - 1));
        float mean = colsum[c] * invN;
        float var  = colsumsq[c] * invN - mean * mean;
        float v = out[idx];
        out[idx] = (v - mean) * rsqrtf(var + 1e-5f) * gamma[c] + beta[c];
    }
}

// ---------------------------------------------------------------------------
extern "C" void kernel_launch(void* const* d_in, const int* in_sizes, int n_in,
                              void* d_out, int out_size, void* d_ws, size_t ws_size,
                              hipStream_t stream) {
    const float* x     = (const float*)d_in[0];
    const int*   nbr   = (const int*)  d_in[1];
    const float* W1    = (const float*)d_in[2];
    const float* b1    = (const float*)d_in[3];
    const float* W2    = (const float*)d_in[4];
    const float* b2    = (const float*)d_in[5];
    const float* gamma = (const float*)d_in[6];
    const float* beta  = (const float*)d_in[7];
    const float* alpha = (const float*)d_in[8];
    const float* eps   = (const float*)d_in[9];

    const int nrows = in_sizes[0] / FEAT;          // 100000

    char* ws = (char*)d_ws;
    const size_t big  = (size_t)nrows * FEAT * sizeof(float);   // 102.4 MB
    const size_t half = (size_t)nrows * FEAT * sizeof(unsigned short);
    // region0: h0 (f32) -> later act1hi + act1lo (bf16 pair, same bytes)
    float*          h0     = (float*)ws;
    unsigned short* a1hi   = (unsigned short*)ws;
    unsigned short* a1lo   = (unsigned short*)(ws + half);
    // region1: h1hi + h1lo
    unsigned short* h1hi   = (unsigned short*)(ws + big);
    unsigned short* h1lo   = (unsigned short*)(ws + big + half);
    // small tail
    unsigned short* w1hi = (unsigned short*)(ws + 2 * big);   // 64K ushort each
    unsigned short* w1lo = w1hi + 65536;
    unsigned short* w2hi = w1lo + 65536;
    unsigned short* w2lo = w2hi + 65536;
    unsigned int*   cmin = (unsigned int*)(w2lo + 65536);
    float*        colsum = (float*)(cmin + FEAT);
    float*         colsq = colsum + FEAT;

    k_wsplit<<<16, 256, 0, stream>>>(W1, w1hi, w1lo);
    k_wsplit<<<16, 256, 0, stream>>>(W2, w2hi, w2lo);
    k_init<<<1, 256, 0, stream>>>(cmin, colsum, colsq);

    k_aggregate<<<nrows, 256, 0, stream>>>(x, nbr, h0);
    k_colmin<<<512, 256, 0, stream>>>(h0, cmin, nrows);
    k_combine<<<nrows, 256, 0, stream>>>(x, nbr, h0, cmin, alpha, eps,
                                         h1hi, h1lo);

    dim3 gg((nrows + 127) / 128, FEAT / 32);
    k_gemm<<<gg, 256, 0, stream>>>(h1hi, h1lo, w1hi, w1lo, b1,
                                   a1hi, a1lo, nullptr, nrows, 0,
                                   nullptr, nullptr);
    k_gemm<<<gg, 256, 0, stream>>>(a1hi, a1lo, w2hi, w2lo, b2,
                                   nullptr, nullptr, (float*)d_out, nrows, 1,
                                   colsum, colsq);

    k_bn<<<4096, 256, 0, stream>>>((float*)d_out, colsum, colsq, gamma, beta,
                                   nrows);
}
